// CSNet_init_146028888176
// MI455X (gfx1250) — matrix operand order repaired and checked by reading, not compiled, not run on hardware
//
#include <hip/hip_runtime.h>

// ---------------------------------------------------------------------------
// CSNet block-compressed-sensing forward, fused for gfx1250 (MI455X).
//   GEMM1: [16384 x 1024] * [1024 x 256]  -> outcsy
//   GEMM2: [16384 x 256]  * [256 x 1024]  -> out, rec (transposed scatter)
// bf16 WMMA (v_wmma_f32_16x16x32_bf16) with f32 accumulation; GEMMs fused
// through LDS so outcsy never round-trips HBM as an operand.
// Workspace: 1 MB (two packed bf16 weight images). Harness ws is >= this.
// ---------------------------------------------------------------------------

typedef __attribute__((ext_vector_type(16))) __bf16 v16bf;
typedef __attribute__((ext_vector_type(8)))  __bf16 v8bf;
typedef __attribute__((ext_vector_type(8)))  float  v8f;

static __device__ __forceinline__ __bf16 f2bf(float f) {
  union { float f; unsigned u; } v; v.f = f;
  unsigned u = v.u + 0x7FFFu + ((v.u >> 16) & 1u);   // round-to-nearest-even
  unsigned short h = (unsigned short)(u >> 16);
  __bf16 b;
  __builtin_memcpy(&b, &h, sizeof(h));
  return b;
}

#define WMMA_BF16(A, B, C) \
  __builtin_amdgcn_wmma_f32_16x16x32_bf16(false, (A), false, (B), (short)0, (C), false, false)

#define SHUF16(LO, HI) \
  __builtin_shufflevector((LO), (HI), 0,1,2,3,4,5,6,7,8,9,10,11,12,13,14,15)

// ---------------------------------------------------------------------------
// Weight packers: emit bf16 in the per-lane WMMA B-fragment order.
// B tile (K=32 x N=16): lane l -> column n = l&15; K group = (l>>4)*8;
// element e (0..15) -> K = (e>>3)*16 + Kgroup + (e&7).  One tile = 32 lanes * 16 bf16 = 1 KB.
// ---------------------------------------------------------------------------

// w_sample: f32 [256 c][1024 k], k = p*32+q.  Tiles: kt in 0..31, nt in 0..15.
__global__ __launch_bounds__(256) void pack_wsample(const float* __restrict__ w,
                                                    __bf16* __restrict__ pw) {
  int idx  = blockIdx.x * 256 + threadIdx.x;   // 0 .. 262143
  int tile = idx >> 9;                          // kt*16 + nt
  int le   = idx & 511;
  int l = le >> 4, e = le & 15;
  int kt = tile >> 4, nt = tile & 15;
  int k = (kt << 5) + ((e >> 3) << 4) + ((l >> 4) << 3) + (e & 7);
  int c = (nt << 4) + (l & 15);
  pw[idx] = f2bf(w[c * 1024 + k]);
}

// w_init: f32 [1024 d][256 c].  Tiles: kt2 in 0..7 (K=c), nt2 in 0..63 (N=d).
__global__ __launch_bounds__(256) void pack_winit(const float* __restrict__ w,
                                                  __bf16* __restrict__ pw) {
  int idx  = blockIdx.x * 256 + threadIdx.x;   // 0 .. 262143
  int tile = idx >> 9;                          // kt2*64 + nt2
  int le   = idx & 511;
  int l = le >> 4, e = le & 15;
  int kt2 = tile >> 6, nt2 = tile & 63;
  int k = (kt2 << 5) + ((e >> 3) << 4) + ((l >> 4) << 3) + (e & 7);     // channel c
  int d = (nt2 << 4) + (l & 15);
  pw[idx] = f2bf(w[d * 256 + k]);
}

// ---------------------------------------------------------------------------
// Fused main kernel. One workgroup = 16 image blocks (M-tile of 16).
// ---------------------------------------------------------------------------
__global__ __launch_bounds__(256)
void csnet_fused(const float*  __restrict__ x,
                 const __bf16* __restrict__ pw1,   // packed w_sample^T, bf16
                 const __bf16* __restrict__ pw2,   // packed w_init^T,  bf16
                 float* __restrict__ rec,
                 float* __restrict__ outcsy,
                 float* __restrict__ out) {
  __shared__ __align__(16) __bf16 sA1[16 * 1024];  // 32 KB: x blocks, bf16
  __shared__ __align__(16) __bf16 sA2[16 * 256];   //  8 KB: GEMM1 result, bf16

  const int tid  = threadIdx.x;
  const int wave = tid >> 5;
  const int lane = tid & 31;
  const int bi0  = blockIdx.x << 4;          // first of 16 consecutive blocks

  // ---- Stage A1: gather 16 blocks of x, convert to bf16, into LDS ----------
#pragma unroll
  for (int r = 0; r < 16; ++r) {
    int idx = r * 256 + tid;                 // 0..4095 float4 slots
    int blk = idx >> 8;                      // block within tile
    int rem = idx & 255;
    int p = rem >> 3, q4 = (rem & 7) << 2;   // row p, col q4 within 32x32
    int bi = bi0 + blk;
    int n = bi >> 10, ij = bi & 1023, i = ij >> 5, j = ij & 31;
    const float4 v = *(const float4*)(x + ((size_t)n << 20) +
                                      (size_t)((i << 5) + p) * 1024 +
                                      (j << 5) + q4);
    __bf16* dst = &sA1[(blk << 10) + (p << 5) + q4];
    dst[0] = f2bf(v.x); dst[1] = f2bf(v.y); dst[2] = f2bf(v.z); dst[3] = f2bf(v.w);
  }
  __syncthreads();

  const int mrow = lane & 15;                // A-fragment row owned by lane
  const int koff = (lane >> 4) << 3;         // A-fragment K interleave: 0 or 8
  const int ncol = lane & 15;                // C/D column owned by lane
  const int mofs = (lane >> 4) << 3;         // C/D row offset: 0 or 8

  // ---- GEMM1: Y[16x256] = A1[16x1024] x Wsample^T ; wave owns 2 N-tiles ----
  v8f acc[2] = {};
  for (int kt = 0; kt < 32; ++kt) {
    const __bf16* ap = &sA1[(mrow << 10) + (kt << 5) + koff];
    v8bf alo = *(const v8bf*)(ap);           // K = koff .. koff+7
    v8bf ahi = *(const v8bf*)(ap + 16);      // K = koff+16 .. koff+23
    v16bf a = SHUF16(alo, ahi);
    if (kt + 1 < 32)                         // warm L2/L0 for next B tiles
      __builtin_prefetch(pw1 + ((((kt + 1) << 4) + (wave << 1)) << 9), 0, 1);
#pragma unroll
    for (int t = 0; t < 2; ++t) {
      int nt = (wave << 1) + t;
      v16bf b = *(const v16bf*)(pw1 + (((kt << 4) + nt) << 9) + (lane << 4));
      acc[t] = WMMA_BF16(a, b, acc[t]);
    }
  }

  // ---- Emit outcsy (f32) and feed Y back as bf16 A2 panel ------------------
#pragma unroll
  for (int t = 0; t < 2; ++t) {
    int c = (((wave << 1) + t) << 4) + ncol;
#pragma unroll
    for (int r = 0; r < 8; ++r) {
      int m = mofs + r;
      int bi = bi0 + m;
      int n = bi >> 10, ij = bi & 1023, i = ij >> 5, j = ij & 31;
      float y = acc[t][r];
      outcsy[((size_t)n << 18) + ((size_t)c << 10) + (i << 5) + j] = y;
      sA2[(m << 8) + c] = f2bf(y);
    }
  }
  __syncthreads();

  // ---- GEMM2: Z[16x1024] = A2[16x256] x Winit^T ; wave owns 8 N-tiles ------
  for (int t = 0; t < 8; ++t) {
    int nt2 = (wave << 3) + t;
    v8f z = {};
#pragma unroll
    for (int kt2 = 0; kt2 < 8; ++kt2) {
      const __bf16* ap = &sA2[(mrow << 8) + (kt2 << 5) + koff];
      v8bf alo = *(const v8bf*)(ap);
      v8bf ahi = *(const v8bf*)(ap + 16);
      v16bf a = SHUF16(alo, ahi);
      v16bf b = *(const v16bf*)(pw2 + (((kt2 << 6) + nt2) << 9) + (lane << 4));
      z = WMMA_BF16(a, b, z);
    }
    int d = (nt2 << 4) + ncol;               // output channel
    int p = d & 31, q = d >> 5;              // rec: channel q*32+p -> (p,q)
#pragma unroll
    for (int r = 0; r < 8; ++r) {
      int m = mofs + r;
      int bi = bi0 + m;
      int n = bi >> 10, ij = bi & 1023, i = ij >> 5, j = ij & 31;
      float zv = z[r];
      out[((size_t)n << 20) + ((size_t)d << 10) + (i << 5) + j] = zv;
      rec[((size_t)n << 20) + (size_t)((i << 5) + p) * 1024 + (j << 5) + q] = zv;
    }
  }
}

// ---------------------------------------------------------------------------
extern "C" void kernel_launch(void* const* d_in, const int* in_sizes, int n_in,
                              void* d_out, int out_size, void* d_ws, size_t ws_size,
                              hipStream_t stream) {
  const float* x  = (const float*)d_in[0];   // 16*1*1024*1024
  const float* ws = (const float*)d_in[1];   // 256*1*32*32
  const float* wi = (const float*)d_in[2];   // 1024*256*1*1

  __bf16* pw1 = (__bf16*)d_ws;               // 262144 bf16 = 512 KB
  __bf16* pw2 = pw1 + 262144;                // 262144 bf16 = 512 KB

  float* rec    = (float*)d_out;             // 16777216
  float* outcsy = rec + 16777216;            //  4194304
  float* out    = outcsy + 4194304;          // 16777216

  pack_wsample<<<dim3(1024), dim3(256), 0, stream>>>(ws, pw1);
  pack_winit  <<<dim3(1024), dim3(256), 0, stream>>>(wi, pw2);
  csnet_fused <<<dim3(1024), dim3(256), 0, stream>>>(x, pw1, pw2, rec, outcsy, out);
}
